// Transformer_24575802868264
// MI455X (gfx1250) — matrix operand order repaired and checked by reading, not compiled, not run on hardware
//
#include <hip/hip_runtime.h>
#include <stdint.h>

typedef __attribute__((ext_vector_type(16))) _Float16 v16h;
typedef __attribute__((ext_vector_type(8)))  float    v8f;
typedef __attribute__((ext_vector_type(4)))  unsigned int v4u;
typedef __attribute__((ext_vector_type(2)))  unsigned int v2u;
typedef __attribute__((ext_vector_type(4)))  int i4_t;
typedef __attribute__((ext_vector_type(2)))  int i2_t;

union FragA { v16h v; uint32_t u[8]; };
union H8 { _Float16 h[8]; v4u u4; };   // 16 bytes
union H4 { _Float16 h[4]; v2u u2; };   // 8 bytes

// ---------- CDNA5 async global->LDS copies (ASYNCcnt), with fallback ----------
#if __has_builtin(__builtin_amdgcn_global_load_async_to_lds_b128) && \
    __has_builtin(__builtin_amdgcn_global_load_async_to_lds_b64)
#define T24_ASYNC 1
#else
#define T24_ASYNC 0
#endif

#if T24_ASYNC
typedef __attribute__((address_space(1))) i4_t g_i4;
typedef __attribute__((address_space(3))) i4_t l_i4;
typedef __attribute__((address_space(1))) i2_t g_i2;
typedef __attribute__((address_space(3))) i2_t l_i2;
#endif

__device__ __forceinline__ void t24_cp16(const _Float16* g, _Float16* l) {
#if T24_ASYNC
  __builtin_amdgcn_global_load_async_to_lds_b128((g_i4*)g, (l_i4*)l, 0, 0);
#else
  *(v4u*)l = *(const v4u*)g;
#endif
}
__device__ __forceinline__ void t24_cp8(const _Float16* g, _Float16* l) {
#if T24_ASYNC
  __builtin_amdgcn_global_load_async_to_lds_b64((g_i2*)g, (l_i2*)l, 0, 0);
#else
  *(v2u*)l = *(const v2u*)g;
#endif
}
__device__ __forceinline__ void t24_async_wait() {
#if T24_ASYNC
  asm volatile("s_wait_asynccnt 0" ::: "memory");
#endif
}

// ============================================================
// im2col -> f16: x (64,1,512,128) fp32 -> patch16 (32768 x 128) f16
// ============================================================
__global__ void t24_im2col(const float* __restrict__ x, _Float16* __restrict__ patch) {
  int i4 = blockIdx.x * 256 + threadIdx.x;       // 32768*128/4 units
  int row = i4 >> 5, kk4 = (i4 & 31) * 4;
  int b = row >> 9, s = row & 511;
  int oy = s >> 4, ox = s & 15;
  int ky = kk4 >> 3, kx = kk4 & 7;
  const float4 v = *(const float4*)&x[((size_t)(b * 512 + oy * 16 + ky)) * 128 + ox * 8 + kx];
  H4 p;
  p.h[0] = (_Float16)v.x; p.h[1] = (_Float16)v.y;
  p.h[2] = (_Float16)v.z; p.h[3] = (_Float16)v.w;
  *(v2u*)&patch[(size_t)i4 * 4] = p.u2;
}

// ============================================================
// tok = conv_out + pos_emb; write fp32 (residual) + f16 (GEMM A) copies
// ============================================================
__global__ void t24_posemb(const float* __restrict__ conv, const float* __restrict__ pe,
                           float* __restrict__ out32, _Float16* __restrict__ out16) {
  int i4 = blockIdx.x * 256 + threadIdx.x;       // 32768*512/4 units
  int row = i4 >> 7;
  int col4 = (i4 & 127) * 4;
  float4 t = *(const float4*)&conv[(size_t)i4 * 4];
  const float4 p = *(const float4*)&pe[(size_t)(row & 511) * 512 + col4];
  t.x += p.x; t.y += p.y; t.z += p.z; t.w += p.w;
  *(float4*)&out32[(size_t)i4 * 4] = t;
  H4 hp;
  hp.h[0] = (_Float16)t.x; hp.h[1] = (_Float16)t.y;
  hp.h[2] = (_Float16)t.z; hp.h[3] = (_Float16)t.w;
  *(v2u*)&out16[(size_t)i4 * 4] = hp.u2;
}

// ============================================================
// Straight fp32 -> f16 convert (conv_w is already N x K)
// ============================================================
__global__ void t24_wcvt(const float* __restrict__ in, _Float16* __restrict__ out, int n4) {
  int i = blockIdx.x * 256 + threadIdx.x;
  if (i < n4) {
    const float4 f = ((const float4*)in)[i];
    H4 p;
    p.h[0] = (_Float16)f.x; p.h[1] = (_Float16)f.y;
    p.h[2] = (_Float16)f.z; p.h[3] = (_Float16)f.w;
    ((v2u*)out)[i] = p.u2;
  }
}

// ============================================================
// Tiled transpose-convert: in (K x N fp32) -> out (N x K f16)
// ============================================================
__global__ __launch_bounds__(256)
void t24_wcvtT(const float* __restrict__ in, _Float16* __restrict__ out, int K, int N) {
  __shared__ float t[32][33];
  const int kb = blockIdx.x * 32, nb = blockIdx.y * 32;
  const int tx = threadIdx.x & 31, ty = threadIdx.x >> 5;
  #pragma unroll
  for (int j = 0; j < 4; ++j) {
    int k = kb + ty + j * 8, n = nb + tx;
    t[ty + j * 8][tx] = (k < K && n < N) ? in[(size_t)k * N + n] : 0.f;
  }
  __syncthreads();
  #pragma unroll
  for (int j = 0; j < 4; ++j) {
    int n = nb + ty + j * 8, k = kb + tx;
    if (n < N && k < K) out[(size_t)n * K + k] = (_Float16)t[tx][ty + j * 8];
  }
}

// ============================================================
// WMMA GEMM: C = A(MxK f16) * W(KxN) [+bias], W as Bt = N x K f16.
// Output: fp32 (C) or f16 (C16), exactly one non-null.
// M%64==0, K%32==0 assumed; N guarded only at the store (B rows past N
// are garbage but feed only masked-out columns; weight arena has slack).
// 256 threads = 8 waves; tile 64(M) x 32(N); K step 32. Async LDS staging.
// ============================================================
__global__ __launch_bounds__(256)
void t24_gemm(const _Float16* __restrict__ A, int lda,
              const _Float16* __restrict__ Bt, int ldb,
              const float* __restrict__ bias,
              float* __restrict__ C, _Float16* __restrict__ C16, int ldc,
              int M, int N, int K) {
  __shared__ __attribute__((aligned(16))) _Float16 Alds[64 * 40];
  __shared__ __attribute__((aligned(16))) _Float16 Blds[32 * 40];

  const int tid  = threadIdx.x;
  const int lane = tid & 31, w = tid >> 5;
  const int lm = lane & 15, half = lane >> 4;
  const int mOff = (w >> 1) * 16;
  const int nOff = (w & 1) * 16;
  const int rowBase = blockIdx.y * 64;
  const int colBase = blockIdx.x * 32;

  const int ar = tid >> 2, ac8 = (tid & 3) * 8;   // A: 64x32, 8 f16/thread
  const int bn = tid >> 3, bc4 = (tid & 7) * 4;   // B: 32x32, 4 f16/thread
  const _Float16* Ag = A + (size_t)(rowBase + ar) * lda + ac8;
  const _Float16* Bg = Bt + (size_t)(colBase + bn) * ldb + bc4;
  _Float16* Al = &Alds[ar * 40 + ac8];
  _Float16* Bl = &Blds[bn * 40 + bc4];

  v8f acc = {0.f,0.f,0.f,0.f,0.f,0.f,0.f,0.f};

  for (int k0 = 0; k0 < K; k0 += 32) {
    __syncthreads();
    t24_cp16(Ag + k0, Al);
    t24_cp8(Bg + k0, Bl);
    t24_async_wait();
    __syncthreads();

    FragA fa, fb;
    #pragma unroll
    for (int vv = 0; vv < 8; ++vv) {
      int kb = ((vv & 3) << 1) + (half << 3) + ((vv >> 2) << 4);
      fa.u[vv] = *(const uint32_t*)&Alds[(mOff + lm) * 40 + kb];
    }
    #pragma unroll
    for (int vv = 0; vv < 8; ++vv) {
      int kb = ((vv & 3) << 1) + (half << 3) + ((vv >> 2) << 4);
      fb.u[vv] = *(const uint32_t*)&Blds[(nOff + lm) * 40 + kb];
    }
    acc = __builtin_amdgcn_wmma_f32_16x16x32_f16(false, fa.v, false, fb.v,
                                                 (short)0, acc, false, false);
  }

  const int gn = colBase + nOff + lm;
  if (gn < N) {
    const float bv = bias ? bias[gn] : 0.f;
    if (C) {
      #pragma unroll
      for (int r = 0; r < 8; ++r) {
        int gm = rowBase + mOff + r + (half << 3);
        C[(size_t)gm * ldc + gn] = acc[r] + bv;
      }
    } else {
      #pragma unroll
      for (int r = 0; r < 8; ++r) {
        int gm = rowBase + mOff + r + (half << 3);
        C16[(size_t)gm * ldc + gn] = (_Float16)(acc[r] + bv);
      }
    }
  }
}

// ============================================================
// Flash attention, all-f16 operands: q,k,v,z (64*512) x 256 f16.
// Score scale 1/sqrt(64) applied to fp32 scores post-WMMA.
// Grid (4 q-blocks of 128, 4 heads, 64 batch); 256 threads = 8 waves.
// ============================================================
__global__ __launch_bounds__(256)
void t24_attn(const _Float16* __restrict__ q, const _Float16* __restrict__ k,
              const _Float16* __restrict__ v, _Float16* __restrict__ z) {
  __shared__ __attribute__((aligned(16))) _Float16 Qs[128 * 72]; // [qrow][d]
  __shared__ __attribute__((aligned(16))) _Float16 Ks[32 * 72];  // [key][d]
  __shared__ __attribute__((aligned(16))) _Float16 Vs[64 * 34];  // [d][key]
  __shared__ __attribute__((aligned(16))) _Float16 Ps[8][16 * 34];

  const int tid  = threadIdx.x;
  const int lane = tid & 31, w = tid >> 5;
  const int lm = lane & 15, half = lane >> 4;
  const int qBase = blockIdx.x * 128;
  const int h = blockIdx.y;
  const int b = blockIdx.z;
  const size_t base = ((size_t)b * 512) * 256 + (size_t)h * 64;

  // Q block: pure async copies (128x64 f16)
  #pragma unroll
  for (int j = 0; j < 4; ++j) {
    int u = tid + j * 256;
    int r = u >> 3, d8 = (u & 7) * 8;
    t24_cp16(q + base + (size_t)(qBase + r) * 256 + d8, &Qs[r * 72 + d8]);
  }
  t24_async_wait();
  __syncthreads();

  const int wOff = w * 16;
  FragA QA[2];
  #pragma unroll
  for (int c = 0; c < 2; ++c) {
    #pragma unroll
    for (int vv = 0; vv < 8; ++vv) {
      int kb = c * 32 + ((vv & 3) << 1) + (half << 3) + ((vv >> 2) << 4);
      QA[c].u[vv] = *(const uint32_t*)&Qs[(wOff + lm) * 72 + kb];
    }
  }

  const v8f zero8 = {0.f,0.f,0.f,0.f,0.f,0.f,0.f,0.f};
  v8f O[4] = {zero8, zero8, zero8, zero8};
  float mrow[8], lrow[8];
  #pragma unroll
  for (int r = 0; r < 8; ++r) { mrow[r] = -1e30f; lrow[r] = 0.f; }

  const int skey = tid >> 3, sd8 = (tid & 7) * 8;  // one 8-f16 unit / thread

  for (int kb0 = 0; kb0 < 512; kb0 += 32) {
    __syncthreads();
    // K chunk: async copy; V chunk: load + transpose scatter
    t24_cp16(k + base + (size_t)(kb0 + skey) * 256 + sd8, &Ks[skey * 72 + sd8]);
    {
      H8 p;
      p.u4 = *(const v4u*)(v + base + (size_t)(kb0 + skey) * 256 + sd8);
      #pragma unroll
      for (int i = 0; i < 8; ++i) Vs[(sd8 + i) * 34 + skey] = p.h[i];
    }
    t24_async_wait();
    __syncthreads();

    // S = Q * K^T (two 16x16 f32 tiles over 32 keys)
    v8f S0 = zero8, S1 = zero8;
    #pragma unroll
    for (int c = 0; c < 2; ++c) {
      FragA fb0, fb1;
      #pragma unroll
      for (int vv = 0; vv < 8; ++vv) {
        int kb = c * 32 + ((vv & 3) << 1) + (half << 3) + ((vv >> 2) << 4);
        fb0.u[vv] = *(const uint32_t*)&Ks[(lm)      * 72 + kb];
        fb1.u[vv] = *(const uint32_t*)&Ks[(16 + lm) * 72 + kb];
      }
      S0 = __builtin_amdgcn_wmma_f32_16x16x32_f16(false, QA[c].v, false, fb0.v, (short)0, S0, false, false);
      S1 = __builtin_amdgcn_wmma_f32_16x16x32_f16(false, QA[c].v, false, fb1.v, (short)0, S1, false, false);
    }

    // Online softmax (rows r + 8*half; reduce across 16 lanes)
    #pragma unroll
    for (int r = 0; r < 8; ++r) {
      float s0 = S0[r] * 0.125f, s1 = S1[r] * 0.125f;
      float mx = fmaxf(s0, s1);
      mx = fmaxf(mx, __shfl_xor(mx, 1));
      mx = fmaxf(mx, __shfl_xor(mx, 2));
      mx = fmaxf(mx, __shfl_xor(mx, 4));
      mx = fmaxf(mx, __shfl_xor(mx, 8));
      float mnew = fmaxf(mrow[r], mx);
      float p0 = __expf(s0 - mnew);
      float p1 = __expf(s1 - mnew);
      float ps = p0 + p1;
      ps += __shfl_xor(ps, 1);
      ps += __shfl_xor(ps, 2);
      ps += __shfl_xor(ps, 4);
      ps += __shfl_xor(ps, 8);
      float alpha = __expf(mrow[r] - mnew);
      lrow[r] = lrow[r] * alpha + ps;
      mrow[r] = mnew;
      #pragma unroll
      for (int t = 0; t < 4; ++t) O[t][r] *= alpha;
      int m = r + (half << 3);
      Ps[w][m * 34 + lm]      = (_Float16)p0;
      Ps[w][m * 34 + 16 + lm] = (_Float16)p1;
    }
    __syncthreads();

    // O += P(16x32) * V(32x64)
    FragA fp;
    #pragma unroll
    for (int vv = 0; vv < 8; ++vv) {
      int kb = ((vv & 3) << 1) + (half << 3) + ((vv >> 2) << 4);
      fp.u[vv] = *(const uint32_t*)&Ps[w][lm * 34 + kb];
    }
    #pragma unroll
    for (int t = 0; t < 4; ++t) {
      FragA fb;
      #pragma unroll
      for (int vv = 0; vv < 8; ++vv) {
        int kb = ((vv & 3) << 1) + (half << 3) + ((vv >> 2) << 4);
        fb.u[vv] = *(const uint32_t*)&Vs[(t * 16 + lm) * 34 + kb];
      }
      O[t] = __builtin_amdgcn_wmma_f32_16x16x32_f16(false, fp.v, false, fb.v, (short)0, O[t], false, false);
    }
  }

  // Finalize: divide by row sum, write f16
  #pragma unroll
  for (int t = 0; t < 4; ++t) {
    #pragma unroll
    for (int r = 0; r < 8; ++r) {
      int gm = qBase + wOff + r + (half << 3);
      z[base + (size_t)gm * 256 + t * 16 + lm] = (_Float16)(O[t][r] / lrow[r]);
    }
  }
}

// ============================================================
// Fused LayerNorm (+optional fp32 residual) + ReLU.
// Writes fp32 and/or f16 copies (either may be null).
// ============================================================
__global__ __launch_bounds__(256)
void t24_ln_act(const float* __restrict__ X, const float* __restrict__ resid,
                const float* __restrict__ g, const float* __restrict__ bta,
                float* __restrict__ out32, _Float16* __restrict__ out16, int D) {
  int row = blockIdx.x;
  const float* xr = X + (size_t)row * D;
  float s = 0.f, ss = 0.f;
  for (int i = threadIdx.x; i < D; i += 256) {
    float v = xr[i];
    s += v; ss += v * v;
  }
  __shared__ float r1[256], r2[256];
  r1[threadIdx.x] = s; r2[threadIdx.x] = ss;
  __syncthreads();
  for (int off = 128; off > 0; off >>= 1) {
    if (threadIdx.x < off) {
      r1[threadIdx.x] += r1[threadIdx.x + off];
      r2[threadIdx.x] += r2[threadIdx.x + off];
    }
    __syncthreads();
  }
  float mean = r1[0] / (float)D;
  float var  = r2[0] / (float)D - mean * mean;
  float rstd = rsqrtf(fmaxf(var, 0.f) + 1e-5f);
  const float* rr = resid ? (resid + (size_t)row * D) : nullptr;
  for (int i = threadIdx.x; i < D; i += 256) {
    float v = (xr[i] - mean) * rstd * g[i] + bta[i];
    if (rr) v += rr[i];
    v = fmaxf(v, 0.f);
    if (out32) out32[(size_t)row * D + i] = v;
    if (out16) out16[(size_t)row * D + i] = (_Float16)v;
  }
}

// ============================================================
// Host orchestration
// ============================================================
static inline void t24_run_gemm(hipStream_t st, const _Float16* A, int lda,
                                const _Float16* Bt, const float* bias,
                                float* C, _Float16* C16, int ldc, int M, int N, int K) {
  dim3 grid((N + 31) / 32, (M + 63) / 64);
  t24_gemm<<<grid, 256, 0, st>>>(A, lda, Bt, K, bias, C, C16, ldc, M, N, K);
}

static inline void t24_run_wcvtT(hipStream_t st, const float* in, _Float16* out, int K, int N) {
  dim3 grid((K + 31) / 32, (N + 31) / 32);
  t24_wcvtT<<<grid, 256, 0, st>>>(in, out, K, N);
}

extern "C" void kernel_launch(void* const* d_in, const int* in_sizes, int n_in,
                              void* d_out, int out_size, void* d_ws, size_t ws_size,
                              hipStream_t stream) {
  (void)in_sizes; (void)n_in; (void)out_size; (void)ws_size;

  const float* x      = (const float*)d_in[0];
  const float* conv_w = (const float*)d_in[1];   // (512,1,16,8) = N x K already
  const float* conv_b = (const float*)d_in[2];
  const float* pe     = (const float*)d_in[3];
  const float* dt_w1  = (const float*)d_in[48];
  const float* dt_b1  = (const float*)d_in[49];
  const float* dt_g   = (const float*)d_in[50];
  const float* dt_bt  = (const float*)d_in[51];
  const float* dt_w2  = (const float*)d_in[52];
  const float* dt_b2  = (const float*)d_in[53];

  const int MTOK = 64 * 512;                 // 32768 token rows
  const size_t TOKSZ = (size_t)MTOK * 512;
  const size_t HSZ   = (size_t)MTOK * 256;

  // fp32 region
  float* ws     = (float*)d_ws;
  float* bufT1  = ws;                 // GEMM fp32 outputs
  float* tok32  = ws + TOKSZ;         // residual copy of block input

  // f16 region
  _Float16* hbase  = (_Float16*)(ws + 2 * TOKSZ);
  _Float16* tok16  = hbase;                    // M x 512 (shrinks)
  _Float16* tokR16 = tok16 + TOKSZ;            // M x 512
  _Float16* q16    = tokR16 + TOKSZ;           // M x 256
  _Float16* k16    = q16 + HSZ;
  _Float16* v16    = k16 + HSZ;
  _Float16* z16    = v16 + HSZ;
  _Float16* patch16 = q16;                     // alias: only live before q16

  // f16 weight arena
  _Float16* wh = z16 + HSZ;
  size_t woff = 0;
  auto nextw = [&](size_t n) { _Float16* p = wh + woff; woff += (n + 7) & ~(size_t)7; return p; };

  const int DIMS[5] = {512, 256, 128, 64, 32};

  // ---- Weight conversion (f16, N x K layout) ----
  _Float16* convW = nextw(512 * 128);
  t24_wcvt<<<(512 * 128 / 4 + 255) / 256, 256, 0, stream>>>(conv_w, convW, 512 * 128 / 4);

  _Float16 *wqT[4], *wkT[4], *wvT[4], *woT[4], *wfT[4];
  for (int i = 0; i < 4; ++i) {
    const int di = DIMS[i], dn = DIMS[i + 1];
    const float* wq = (const float*)d_in[4 + 11 * i + 0];
    const float* wk = (const float*)d_in[4 + 11 * i + 1];
    const float* wv = (const float*)d_in[4 + 11 * i + 2];
    const float* wo = (const float*)d_in[4 + 11 * i + 3];
    const float* wf = (const float*)d_in[4 + 11 * i + 7];
    wqT[i] = nextw((size_t)256 * di); t24_run_wcvtT(stream, wq, wqT[i], di, 256);
    wkT[i] = nextw((size_t)256 * di); t24_run_wcvtT(stream, wk, wkT[i], di, 256);
    wvT[i] = nextw((size_t)256 * di); t24_run_wcvtT(stream, wv, wvT[i], di, 256);
    woT[i] = nextw((size_t)256 * di); t24_run_wcvtT(stream, wo, woT[i], 256, di);
    wfT[i] = nextw((size_t)di * dn);  t24_run_wcvtT(stream, wf, wfT[i], di, dn);
  }
  _Float16* w1T = nextw((size_t)16384 * 1024); t24_run_wcvtT(stream, dt_w1, w1T, 16384, 1024);
  _Float16* w2T = nextw((size_t)1024 * 8);     t24_run_wcvtT(stream, dt_w2, w2T, 1024, 8);
  (void)nextw((size_t)32 * 1024);              // slack: staging may read past N=8 rows

  // ---- 1) conv stem (non-overlapping patches) ----
  t24_im2col<<<(MTOK * 128 / 4) / 256, 256, 0, stream>>>(x, patch16);
  t24_run_gemm(stream, patch16, 128, convW, conv_b, bufT1, nullptr, 512, MTOK, 512, 128);
  t24_posemb<<<(MTOK * 512 / 4) / 256, 256, 0, stream>>>(bufT1, pe, tok32, tok16);

  // ---- 2) transformer blocks ----
  for (int i = 0; i < 4; ++i) {
    const float* bo = (const float*)d_in[4 + 11 * i + 4];
    const float* g1 = (const float*)d_in[4 + 11 * i + 5];
    const float* b1 = (const float*)d_in[4 + 11 * i + 6];
    const float* bf = (const float*)d_in[4 + 11 * i + 8];
    const float* g2 = (const float*)d_in[4 + 11 * i + 9];
    const float* b2 = (const float*)d_in[4 + 11 * i + 10];
    const int di = DIMS[i], dn = DIMS[i + 1];

    t24_run_gemm(stream, tok16, di, wqT[i], nullptr, nullptr, q16, 256, MTOK, 256, di);
    t24_run_gemm(stream, tok16, di, wkT[i], nullptr, nullptr, k16, 256, MTOK, 256, di);
    t24_run_gemm(stream, tok16, di, wvT[i], nullptr, nullptr, v16, 256, MTOK, 256, di);

    t24_attn<<<dim3(4, 4, 64), 256, 0, stream>>>(q16, k16, v16, z16);

    t24_run_gemm(stream, z16, 256, woT[i], bo, bufT1, nullptr, di, MTOK, di, 256);
    t24_ln_act<<<MTOK, 256, 0, stream>>>(bufT1, tok32, g1, b1, nullptr, tokR16, di);

    t24_run_gemm(stream, tokR16, di, wfT[i], bf, bufT1, nullptr, dn, MTOK, dn, di);
    t24_ln_act<<<MTOK, 256, 0, stream>>>(bufT1, nullptr, g2, b2, tok32, tok16, dn);
  }

  // ---- 3) head: flat (64 x 16384) f16 is exactly tok16 row-major ----
  t24_run_gemm(stream, tok16, 16384, w1T, dt_b1, bufT1, nullptr, 1024, 64, 1024, 16384);
  t24_ln_act<<<64, 256, 0, stream>>>(bufT1, nullptr, dt_g, dt_bt, nullptr, tokR16, 1024);
  t24_run_gemm(stream, tokR16, 1024, w2T, dt_b2, (float*)d_out, nullptr, 8, 64, 8, 1024);
}
